// PLELayer_35991825940841
// MI455X (gfx1250) — matrix operand order repaired and checked by reading, not compiled, not run on hardware
//
#include <hip/hip_runtime.h>
#include <hip/hip_bf16.h>

// ---------------------------------------------------------------------------
// Problem constants (match reference)
// ---------------------------------------------------------------------------
constexpr int B_   = 16384;
constexpr int F_   = 512;
constexpr int T_   = 4;
constexpr int EPT_ = 2;
constexpr int S_   = 2;
constexpr int E_   = 256;
constexpr int H_   = 128;
constexpr int NEXP = T_ * EPT_ + S_;   // 10 experts per level
constexpr int G0   = EPT_ + S_;        // 4 gates per task
constexpr int GS   = T_ * EPT_ + S_;   // 10 shared gates (level 0)

// ---------------------------------------------------------------------------
// WMMA types (CDNA5 gfx1250, wave32)
// ---------------------------------------------------------------------------
typedef __attribute__((ext_vector_type(16))) __bf16 bf16x16;
typedef __attribute__((ext_vector_type(8)))  __bf16 bf16x8;
typedef __attribute__((ext_vector_type(8)))  float  f32x8;

union AFrag { bf16x16 v; bf16x8 h[2]; };

__device__ inline f32x8 zero8() {
    f32x8 z;
#pragma unroll
    for (int i = 0; i < 8; ++i) z[i] = 0.f;
    return z;
}

__device__ inline f32x8 wmma_bf16(bf16x16 a, bf16x16 b, f32x8 c) {
    return __builtin_amdgcn_wmma_f32_16x16x32_bf16(
        /*neg_a=*/false, a, /*neg_b=*/false, b,
        /*c_mod=*/(short)0, c, /*reuse_a=*/false, /*reuse_b=*/false);
}

// ---------------------------------------------------------------------------
// Prep: f32 -> bf16 copy
// ---------------------------------------------------------------------------
__global__ void f32_to_bf16_kernel(const float* __restrict__ src,
                                   __bf16* __restrict__ dst, int n) {
    int i = blockIdx.x * blockDim.x + threadIdx.x;
    if (i < n) dst[i] = (__bf16)src[i];
}

// ---------------------------------------------------------------------------
// Prep: swizzle weight matrices [J][K][N] f32 into B-fragment-native bf16:
//   dst[j][k>>5][n>>4][ ((k>>3)&1)*16 + (n&15) ][ ((k>>4)&1)*8 + (k&7) ]
// One lane's B fragment (V_WMMA_F32_16X16X32_BF16, 32x16 B tile) is then a
// single contiguous 32-byte load.
// ---------------------------------------------------------------------------
__global__ void swizzle_w_kernel(const float* __restrict__ src,
                                 __bf16* __restrict__ dst,
                                 int J, int K, int N) {
    int i = blockIdx.x * blockDim.x + threadIdx.x;
    int total = J * K * N;
    if (i >= total) return;
    int j   = i / (K * N);
    int rem = i - j * (K * N);
    int k   = rem / N;
    int n   = rem - k * N;
    int tile = (k >> 5) * (N >> 4) + (n >> 4);
    int lane = (((k >> 3) & 1) << 4) | (n & 15);
    int el   = (((k >> 4) & 1) << 3) | (k & 7);
    dst[(size_t)j * K * N + (size_t)tile * 512 + lane * 16 + el] = (__bf16)src[i];
}

// ---------------------------------------------------------------------------
// Level-0 expert GEMMs: exp[j] = relu(x @ W[j] + b[j]), j = 0..9
// Block: 256 threads (8 waves).  M-tile = 32 rows of x staged ONCE in LDS,
// then looped over all 10 experts.  Each wave owns a 32-column strip and
// computes a 32x32 tile per expert: 2 A-frags x 2 B-frags -> 4 WMMAs/k-step
// (1.0 WMMA per b128 load).  K = 512, N = 256.
// ---------------------------------------------------------------------------
__global__ void __launch_bounds__(256)
gemm_level0_kernel(const __bf16* __restrict__ xb,
                   const __bf16* __restrict__ Wsw,
                   const float* __restrict__ b_task,
                   const float* __restrict__ b_sh,
                   __bf16* __restrict__ expo) {
    constexpr int MT = 32, K = F_, N = E_, LDA = K + 8;  // +8 elems => conflict-free frag reads
    __shared__ __bf16 ldsA[MT * LDA];

    const int tid   = threadIdx.x;
    const int Mbase = blockIdx.x * MT;

    // Stage A tile (bf16, 16-byte vector copies)
    {
        const uint4* src = (const uint4*)(xb + (size_t)Mbase * K);
        for (int i = tid; i < MT * (K / 8); i += 256) {
            int r = i >> 6, c = i & 63;                       // K/8 = 64 chunks/row
            *(uint4*)&ldsA[r * LDA + c * 8] = src[(size_t)r * (K / 8) + c];
        }
    }
    __syncthreads();

    const int w = tid >> 5, lane = tid & 31;                  // wave w owns cols [w*32, w*32+32)
    const int arow   = lane & 15;
    const int kbase  = (lane >> 4) * 8;                       // A-matrix 16x32 bf16 layout
    const int rshift = (lane >> 4) * 8;                       // C/D layout: M = r + 8*(lane>=16)
    const int ncol   = lane & 15;

    for (int j = 0; j < NEXP; ++j) {
        const __bf16* Wj = Wsw + (size_t)j * K * N;
        if (j + 1 < NEXP) __builtin_prefetch(Wsw + (size_t)(j + 1) * K * N, 0, 1);
        const float* bias = (j < 8) ? (b_task + j * N) : (b_sh + (j - 8) * N);
        __bf16* outj = expo + (size_t)j * B_ * N;

        f32x8 acc[2][2];
#pragma unroll
        for (int mf = 0; mf < 2; ++mf)
#pragma unroll
            for (int nf = 0; nf < 2; ++nf) acc[mf][nf] = zero8();

#pragma unroll 2
        for (int k0 = 0; k0 < K / 32; ++k0) {
            AFrag a0, a1;
            const __bf16* ap0 = &ldsA[arow * LDA + k0 * 32 + kbase];
            const __bf16* ap1 = &ldsA[(arow + 16) * LDA + k0 * 32 + kbase];
            a0.h[0] = *(const bf16x8*)ap0;
            a0.h[1] = *(const bf16x8*)(ap0 + 16);
            a1.h[0] = *(const bf16x8*)ap1;
            a1.h[1] = *(const bf16x8*)(ap1 + 16);
            bf16x16 b0 = *(const bf16x16*)(Wj + ((size_t)(k0 * (N / 16) + w * 2 + 0) * 32 + lane) * 16);
            bf16x16 b1 = *(const bf16x16*)(Wj + ((size_t)(k0 * (N / 16) + w * 2 + 1) * 32 + lane) * 16);
            acc[0][0] = wmma_bf16(a0.v, b0, acc[0][0]);
            acc[0][1] = wmma_bf16(a0.v, b1, acc[0][1]);
            acc[1][0] = wmma_bf16(a1.v, b0, acc[1][0]);
            acc[1][1] = wmma_bf16(a1.v, b1, acc[1][1]);
        }

#pragma unroll
        for (int mf = 0; mf < 2; ++mf) {
#pragma unroll
            for (int nf = 0; nf < 2; ++nf) {
                int col = w * 32 + nf * 16 + ncol;
                float bv = bias[col];
#pragma unroll
                for (int r = 0; r < 8; ++r) {
                    int row = Mbase + mf * 16 + rshift + r;
                    outj[(size_t)row * N + col] = (__bf16)fmaxf(acc[mf][nf][r] + bv, 0.f);
                }
            }
        }
    }
}

// ---------------------------------------------------------------------------
// Level-1 expert GEMMs: exp[j] = relu(A_j @ W[j] + b[j]), A_j = t1[task] or sh1
// Block: M-tile = 64, grid.z = expert.  Waves 2(M) x 4(N), 32x32 wave tiles,
// N-halves looped.  K = 256, N = 256.
// ---------------------------------------------------------------------------
__global__ void __launch_bounds__(256)
gemm_level1_kernel(const __bf16* __restrict__ t1,
                   const __bf16* __restrict__ sh1,
                   const __bf16* __restrict__ Wsw,
                   const float* __restrict__ b_task,
                   const float* __restrict__ b_sh,
                   __bf16* __restrict__ expo) {
    constexpr int MT = 64, K = E_, N = E_, LDA = K + 8;
    __shared__ __bf16 ldsA[MT * LDA];

    const int tid   = threadIdx.x;
    const int Mbase = blockIdx.x * MT;
    const int j     = blockIdx.z;

    const __bf16* A = (j < 8) ? (t1 + (size_t)(j >> 1) * B_ * K) : sh1;
    const float* bias = (j < 8) ? (b_task + j * N) : (b_sh + (j - 8) * N);
    const __bf16* Wj = Wsw + (size_t)j * K * N;
    __bf16* outj = expo + (size_t)j * B_ * N;

    {
        const uint4* src = (const uint4*)(A + (size_t)Mbase * K);
        for (int i = tid; i < MT * (K / 8); i += 256) {
            int r = i >> 5, c = i & 31;                       // K/8 = 32 chunks/row
            *(uint4*)&ldsA[r * LDA + c * 8] = src[(size_t)r * (K / 8) + c];
        }
    }
    __syncthreads();

    const int w = tid >> 5, lane = tid & 31;
    const int wm = w & 1, wn = w >> 1;                        // 2 (M) x 4 (N) wave grid
    const int arow   = wm * 32 + (lane & 15);
    const int kbase  = (lane >> 4) * 8;
    const int rshift = (lane >> 4) * 8;
    const int ncol   = lane & 15;

    for (int nblk = 0; nblk < 2; ++nblk) {
        f32x8 acc[2][2];
#pragma unroll
        for (int mf = 0; mf < 2; ++mf)
#pragma unroll
            for (int nf = 0; nf < 2; ++nf) acc[mf][nf] = zero8();
        const int nb0 = nblk * 8 + wn * 2;

#pragma unroll 2
        for (int k0 = 0; k0 < K / 32; ++k0) {
            AFrag a0, a1;
            const __bf16* ap0 = &ldsA[arow * LDA + k0 * 32 + kbase];
            const __bf16* ap1 = &ldsA[(arow + 16) * LDA + k0 * 32 + kbase];
            a0.h[0] = *(const bf16x8*)ap0;
            a0.h[1] = *(const bf16x8*)(ap0 + 16);
            a1.h[0] = *(const bf16x8*)ap1;
            a1.h[1] = *(const bf16x8*)(ap1 + 16);
            bf16x16 b0 = *(const bf16x16*)(Wj + ((size_t)(k0 * (N / 16) + nb0 + 0) * 32 + lane) * 16);
            bf16x16 b1 = *(const bf16x16*)(Wj + ((size_t)(k0 * (N / 16) + nb0 + 1) * 32 + lane) * 16);
            acc[0][0] = wmma_bf16(a0.v, b0, acc[0][0]);
            acc[0][1] = wmma_bf16(a0.v, b1, acc[0][1]);
            acc[1][0] = wmma_bf16(a1.v, b0, acc[1][0]);
            acc[1][1] = wmma_bf16(a1.v, b1, acc[1][1]);
        }

#pragma unroll
        for (int mf = 0; mf < 2; ++mf) {
#pragma unroll
            for (int nf = 0; nf < 2; ++nf) {
                int col = nblk * 128 + wn * 32 + nf * 16 + ncol;
                float bv = bias[col];
#pragma unroll
                for (int r = 0; r < 8; ++r) {
                    int row = Mbase + wm * 32 + mf * 16 + rshift + r;
                    outj[(size_t)row * N + col] = (__bf16)fmaxf(acc[mf][nf][r] + bv, 0.f);
                }
            }
        }
    }
}

// ---------------------------------------------------------------------------
// Level-0 gates: g0[b] = [softmax4(x@w_gate0[t]+b) x4 tasks | softmax10(x@w_gsh0+b)]
// ---------------------------------------------------------------------------
__global__ void __launch_bounds__(256)
gates0_kernel(const float* __restrict__ x,
              const float* __restrict__ w_gate0, const float* __restrict__ b_gate0,
              const float* __restrict__ w_gsh0,  const float* __restrict__ b_gsh0,
              float* __restrict__ g0) {
    int b = blockIdx.x * blockDim.x + threadIdx.x;
    const float* xr = x + (size_t)b * F_;
    float at[T_][G0];
    float ash[GS];
#pragma unroll
    for (int t = 0; t < T_; ++t)
#pragma unroll
        for (int g = 0; g < G0; ++g) at[t][g] = 0.f;
#pragma unroll
    for (int g = 0; g < GS; ++g) ash[g] = 0.f;

    for (int k = 0; k < F_; ++k) {
        float xv = xr[k];
#pragma unroll
        for (int t = 0; t < T_; ++t) {
            const float* wg = w_gate0 + ((size_t)t * F_ + k) * G0;
#pragma unroll
            for (int g = 0; g < G0; ++g) at[t][g] += xv * wg[g];
        }
        const float* ws = w_gsh0 + (size_t)k * GS;
#pragma unroll
        for (int g = 0; g < GS; ++g) ash[g] += xv * ws[g];
    }

    float* go = g0 + (size_t)b * (T_ * G0 + GS);
#pragma unroll
    for (int t = 0; t < T_; ++t) {
        float v[G0], m = -1e30f, s = 0.f;
#pragma unroll
        for (int g = 0; g < G0; ++g) { v[g] = at[t][g] + b_gate0[t * G0 + g]; m = fmaxf(m, v[g]); }
#pragma unroll
        for (int g = 0; g < G0; ++g) { v[g] = __expf(v[g] - m); s += v[g]; }
        float inv = 1.f / s;
#pragma unroll
        for (int g = 0; g < G0; ++g) go[t * G0 + g] = v[g] * inv;
    }
    {
        float v[GS], m = -1e30f, s = 0.f;
#pragma unroll
        for (int g = 0; g < GS; ++g) { v[g] = ash[g] + b_gsh0[g]; m = fmaxf(m, v[g]); }
#pragma unroll
        for (int g = 0; g < GS; ++g) { v[g] = __expf(v[g] - m); s += v[g]; }
        float inv = 1.f / s;
#pragma unroll
        for (int g = 0; g < GS; ++g) go[T_ * G0 + g] = v[g] * inv;
    }
}

// ---------------------------------------------------------------------------
// Level-1 gates: g1[b][t*4+g] = softmax4(t1[t,b] @ w_gate1[t] + b_gate1[t])
// ---------------------------------------------------------------------------
__global__ void __launch_bounds__(256)
gates1_kernel(const __bf16* __restrict__ t1,
              const float* __restrict__ w_gate1, const float* __restrict__ b_gate1,
              float* __restrict__ g1) {
    int b = blockIdx.x * blockDim.x + threadIdx.x;
    float* go = g1 + (size_t)b * (T_ * G0);
#pragma unroll
    for (int t = 0; t < T_; ++t) {
        const __bf16* ar = t1 + ((size_t)t * B_ + b) * E_;
        float ac[G0] = {0.f, 0.f, 0.f, 0.f};
        for (int k = 0; k < E_; ++k) {
            float xv = (float)ar[k];
            const float* wg = w_gate1 + ((size_t)t * E_ + k) * G0;
#pragma unroll
            for (int g = 0; g < G0; ++g) ac[g] += xv * wg[g];
        }
        float m = -1e30f, s = 0.f;
#pragma unroll
        for (int g = 0; g < G0; ++g) { ac[g] += b_gate1[t * G0 + g]; m = fmaxf(m, ac[g]); }
#pragma unroll
        for (int g = 0; g < G0; ++g) { ac[g] = __expf(ac[g] - m); s += ac[g]; }
        float inv = 1.f / s;
#pragma unroll
        for (int g = 0; g < G0; ++g) go[t * G0 + g] = ac[g] * inv;
    }
}

// ---------------------------------------------------------------------------
// Level-0 combine: gated mixtures -> t1[t,b,:] (4 tasks) and sh1[b,:]
// ---------------------------------------------------------------------------
__global__ void __launch_bounds__(256)
combine0_kernel(const __bf16* __restrict__ expo, const float* __restrict__ g0,
                __bf16* __restrict__ t1, __bf16* __restrict__ sh1) {
    int idx = blockIdx.x * blockDim.x + threadIdx.x;   // B_ * (E_/4) threads
    int b = idx >> 6;
    int e = (idx & 63) * 4;
    float ex[NEXP][4];
#pragma unroll
    for (int j = 0; j < NEXP; ++j) {
        const __bf16* p = expo + ((size_t)j * B_ + b) * E_ + e;
#pragma unroll
        for (int c = 0; c < 4; ++c) ex[j][c] = (float)p[c];
    }
    const float* g = g0 + (size_t)b * (T_ * G0 + GS);
#pragma unroll
    for (int t = 0; t < T_; ++t) {
        float o[4] = {0.f, 0.f, 0.f, 0.f};
#pragma unroll
        for (int gi = 0; gi < G0; ++gi) {
            int j = (gi < EPT_) ? (EPT_ * t + gi) : (T_ * EPT_ + gi - EPT_);
            float w = g[t * G0 + gi];
#pragma unroll
            for (int c = 0; c < 4; ++c) o[c] += w * ex[j][c];
        }
        __bf16* p = t1 + ((size_t)t * B_ + b) * E_ + e;
#pragma unroll
        for (int c = 0; c < 4; ++c) p[c] = (__bf16)o[c];
    }
    {
        float o[4] = {0.f, 0.f, 0.f, 0.f};
#pragma unroll
        for (int j = 0; j < GS; ++j) {
            float w = g[T_ * G0 + j];
#pragma unroll
            for (int c = 0; c < 4; ++c) o[c] += w * ex[j][c];
        }
        __bf16* p = sh1 + (size_t)b * E_ + e;
#pragma unroll
        for (int c = 0; c < 4; ++c) p[c] = (__bf16)o[c];
    }
}

// ---------------------------------------------------------------------------
// Level-1 combine: t2[t,b,:] = sum_g g1[t,g] * exp1[jmap(t,g),b,:]
// ---------------------------------------------------------------------------
__global__ void __launch_bounds__(256)
combine1_kernel(const __bf16* __restrict__ expo, const float* __restrict__ g1,
                __bf16* __restrict__ t2) {
    int idx = blockIdx.x * blockDim.x + threadIdx.x;
    int b = idx >> 6;
    int e = (idx & 63) * 4;
    float ex[NEXP][4];
#pragma unroll
    for (int j = 0; j < NEXP; ++j) {
        const __bf16* p = expo + ((size_t)j * B_ + b) * E_ + e;
#pragma unroll
        for (int c = 0; c < 4; ++c) ex[j][c] = (float)p[c];
    }
    const float* g = g1 + (size_t)b * (T_ * G0);
#pragma unroll
    for (int t = 0; t < T_; ++t) {
        float o[4] = {0.f, 0.f, 0.f, 0.f};
#pragma unroll
        for (int gi = 0; gi < G0; ++gi) {
            int j = (gi < EPT_) ? (EPT_ * t + gi) : (T_ * EPT_ + gi - EPT_);
            float w = g[t * G0 + gi];
#pragma unroll
            for (int c = 0; c < 4; ++c) o[c] += w * ex[j][c];
        }
        __bf16* p = t2 + ((size_t)t * B_ + b) * E_ + e;
#pragma unroll
        for (int c = 0; c < 4; ++c) p[c] = (__bf16)o[c];
    }
}

// ---------------------------------------------------------------------------
// Tower: h = relu(t2[t] @ w_tw[t] + b_tw[t]); logits = h @ w_out[t] + b_out[t];
// softmax(2); clip; write d_out.  WMMA GEMM + fused epilogue via LDS reuse.
// Waves 2(M) x 4(N), 32x32 wave tiles over a 64x128 block tile.
// ---------------------------------------------------------------------------
union TowerSmem {
    __bf16 a[64 * (E_ + 8)];     // 33792 bytes: staged A tile
    float  h[64 * (H_ + 4)];     // 33792 bytes: relu(h) matrix
};

__global__ void __launch_bounds__(256)
tower_kernel(const __bf16* __restrict__ t2, const __bf16* __restrict__ Wsw,
             const float* __restrict__ b_tw,
             const float* __restrict__ w_out, const float* __restrict__ b_out,
             float* __restrict__ out) {
    constexpr int MT = 64, K = E_, N = H_, LDA = K + 8, LDH = N + 4;
    __shared__ TowerSmem sm;

    const int tid   = threadIdx.x;
    const int Mbase = blockIdx.x * MT;
    const int task  = blockIdx.z;

    const __bf16* A  = t2 + (size_t)task * B_ * K;
    const __bf16* Wj = Wsw + (size_t)task * K * N;
    const float* bias = b_tw + task * N;

    {
        const uint4* src = (const uint4*)(A + (size_t)Mbase * K);
        for (int i = tid; i < MT * (K / 8); i += 256) {
            int r = i >> 5, c = i & 31;
            *(uint4*)&sm.a[r * LDA + c * 8] = src[(size_t)r * (K / 8) + c];
        }
    }
    __syncthreads();

    const int w = tid >> 5, lane = tid & 31;
    const int wm = w & 1, wn = w >> 1;        // 2 (M) x 4 (N): 64 x 128 block tile
    const int arow   = wm * 32 + (lane & 15);
    const int kbase  = (lane >> 4) * 8;
    const int rshift = (lane >> 4) * 8;
    const int ncol   = lane & 15;

    f32x8 acc[2][2];
#pragma unroll
    for (int mf = 0; mf < 2; ++mf)
#pragma unroll
        for (int nf = 0; nf < 2; ++nf) acc[mf][nf] = zero8();

#pragma unroll 2
    for (int k0 = 0; k0 < K / 32; ++k0) {
        AFrag a0, a1;
        const __bf16* ap0 = &sm.a[arow * LDA + k0 * 32 + kbase];
        const __bf16* ap1 = &sm.a[(arow + 16) * LDA + k0 * 32 + kbase];
        a0.h[0] = *(const bf16x8*)ap0;
        a0.h[1] = *(const bf16x8*)(ap0 + 16);
        a1.h[0] = *(const bf16x8*)ap1;
        a1.h[1] = *(const bf16x8*)(ap1 + 16);
        bf16x16 b0 = *(const bf16x16*)(Wj + ((size_t)(k0 * (N / 16) + wn * 2 + 0) * 32 + lane) * 16);
        bf16x16 b1 = *(const bf16x16*)(Wj + ((size_t)(k0 * (N / 16) + wn * 2 + 1) * 32 + lane) * 16);
        acc[0][0] = wmma_bf16(a0.v, b0, acc[0][0]);
        acc[0][1] = wmma_bf16(a0.v, b1, acc[0][1]);
        acc[1][0] = wmma_bf16(a1.v, b0, acc[1][0]);
        acc[1][1] = wmma_bf16(a1.v, b1, acc[1][1]);
    }
    __syncthreads();   // everyone done reading sm.a before overwriting as sm.h

#pragma unroll
    for (int mf = 0; mf < 2; ++mf) {
#pragma unroll
        for (int nf = 0; nf < 2; ++nf) {
            int col = wn * 32 + nf * 16 + ncol;
            float bv = bias[col];
#pragma unroll
            for (int r = 0; r < 8; ++r) {
                int rl = wm * 32 + mf * 16 + rshift + r;
                sm.h[rl * LDH + col] = fmaxf(acc[mf][nf][r] + bv, 0.f);
            }
        }
    }
    __syncthreads();

    if (tid < MT) {
        const float* wo = w_out + (size_t)task * N * 2;
        float s0 = 0.f, s1 = 0.f;
        for (int k = 0; k < N; ++k) {
            float hv = sm.h[tid * LDH + k];
            s0 += hv * wo[k * 2 + 0];
            s1 += hv * wo[k * 2 + 1];
        }
        float l0 = s0 + b_out[task * 2 + 0];
        float l1 = s1 + b_out[task * 2 + 1];
        float m  = fmaxf(l0, l1);
        float e0 = __expf(l0 - m), e1 = __expf(l1 - m);
        float inv = 1.f / (e0 + e1);
        float p0 = fminf(fmaxf(e0 * inv, 1e-15f), 1.f - 1e-15f);
        float p1 = fminf(fmaxf(e1 * inv, 1e-15f), 1.f - 1e-15f);
        size_t row = (size_t)task * B_ + (Mbase + tid);
        out[row * 2 + 0] = p0;
        out[row * 2 + 1] = p1;
    }
}

// ---------------------------------------------------------------------------
// Launch
// ---------------------------------------------------------------------------
extern "C" void kernel_launch(void* const* d_in, const int* in_sizes, int n_in,
                              void* d_out, int out_size, void* d_ws, size_t ws_size,
                              hipStream_t stream) {
    const float* x        = (const float*)d_in[0];
    const float* w_task0  = (const float*)d_in[1];
    const float* b_task0  = (const float*)d_in[2];
    const float* w_sh0    = (const float*)d_in[3];
    const float* b_sh0    = (const float*)d_in[4];
    const float* w_gate0  = (const float*)d_in[5];
    const float* b_gate0  = (const float*)d_in[6];
    const float* w_gsh0   = (const float*)d_in[7];
    const float* b_gsh0   = (const float*)d_in[8];
    const float* w_task1  = (const float*)d_in[9];
    const float* b_task1  = (const float*)d_in[10];
    const float* w_sh1    = (const float*)d_in[11];
    const float* b_sh1    = (const float*)d_in[12];
    const float* w_gate1  = (const float*)d_in[13];
    const float* b_gate1  = (const float*)d_in[14];
    const float* w_tw     = (const float*)d_in[15];
    const float* b_tw     = (const float*)d_in[16];
    const float* w_out    = (const float*)d_in[17];
    const float* b_out    = (const float*)d_in[18];

    char* p = (char*)d_ws;
    auto alloc = [&](size_t bytes) -> char* {
        char* r = p;
        p += (bytes + 255) & ~(size_t)255;
        return r;
    };

    __bf16* x_bf  = (__bf16*)alloc((size_t)B_ * F_ * 2);
    __bf16* Wsw0  = (__bf16*)alloc((size_t)NEXP * F_ * E_ * 2);
    __bf16* Wsw1  = (__bf16*)alloc((size_t)NEXP * E_ * E_ * 2);
    __bf16* WswT  = (__bf16*)alloc((size_t)T_ * E_ * H_ * 2);
    __bf16* expb  = (__bf16*)alloc((size_t)NEXP * B_ * E_ * 2); // reused level0 -> level1
    float*  g0    = (float*) alloc((size_t)B_ * (T_ * G0 + GS) * 4);
    __bf16* t1b   = (__bf16*)alloc((size_t)T_ * B_ * E_ * 2);
    __bf16* sh1b  = (__bf16*)alloc((size_t)B_ * E_ * 2);
    float*  g1    = (float*) alloc((size_t)B_ * (T_ * G0) * 4);
    __bf16* t2b   = (__bf16*)alloc((size_t)T_ * B_ * E_ * 2);

    // --- prep: convert x, swizzle all weights into B-fragment layout ---
    {
        int n = B_ * F_;
        f32_to_bf16_kernel<<<(n + 255) / 256, 256, 0, stream>>>(x, x_bf, n);
    }
    auto swz = [&](const float* src, __bf16* dst, int J, int K, int N) {
        int n = J * K * N;
        swizzle_w_kernel<<<(n + 255) / 256, 256, 0, stream>>>(src, dst, J, K, N);
    };
    swz(w_task0, Wsw0,                          8, F_, E_);
    swz(w_sh0,   Wsw0 + (size_t)8 * F_ * E_,    2, F_, E_);
    swz(w_task1, Wsw1,                          8, E_, E_);
    swz(w_sh1,   Wsw1 + (size_t)8 * E_ * E_,    2, E_, E_);
    swz(w_tw,    WswT,                          4, E_, H_);

    // --- level 0 ---
    gemm_level0_kernel<<<dim3(B_ / 32, 1, 1), 256, 0, stream>>>(
        x_bf, Wsw0, b_task0, b_sh0, expb);
    gates0_kernel<<<B_ / 256, 256, 0, stream>>>(
        x, w_gate0, b_gate0, w_gsh0, b_gsh0, g0);
    combine0_kernel<<<(B_ * (E_ / 4)) / 256, 256, 0, stream>>>(
        expb, g0, t1b, sh1b);

    // --- level 1 (expert buffer reused; stream order guarantees safety) ---
    gemm_level1_kernel<<<dim3(B_ / 64, 1, NEXP), 256, 0, stream>>>(
        t1b, sh1b, Wsw1, b_task1, b_sh1, expb);
    gates1_kernel<<<B_ / 256, 256, 0, stream>>>(
        t1b, w_gate1, b_gate1, g1);
    combine1_kernel<<<(B_ * (E_ / 4)) / 256, 256, 0, stream>>>(
        expb, g1, t2b);

    // --- towers + softmax + clip ---
    tower_kernel<<<dim3(B_ / 64, 1, T_), 256, 0, stream>>>(
        t2b, WswT, b_tw, w_out, b_out, (float*)d_out);
}
